// INR_26302379721259
// MI455X (gfx1250) — compile-verified, matrix operand order checked by reading, not run on hardware
//
#include <hip/hip_runtime.h>

// ---------------- problem constants ----------------
#define MAP      128        // fourier mapping size
#define FF_DIM   256        // 2*MAP
#define HID      512
#define NHID     5
#define OUTF     3
#define OMEGA0   30.0f
#define TWO_PI_F 6.28318530717958647692f
#define L_TOTAL  262144
#define MTILE    64         // rows per workgroup
#define THREADS  256        // 8 waves (wave32)

typedef __attribute__((ext_vector_type(16))) __bf16 v16bf;
typedef __attribute__((ext_vector_type(8)))  float  v8f;

// swizzled-weight workspace layout (in __bf16 elements):
//   layer0 : 32 nb * 8 kb fragments  (512 elems = 1 KB each)
//   hidden : 5 layers * 32 nb * 16 kb fragments
#define L0_FRAGS  (32 * 8)
#define HID_FRAGS (32 * 16)
#define TOT_FRAGS (L0_FRAGS + NHID * HID_FRAGS)
#define L0_ELEMS  (L0_FRAGS * 512)
#define HID_ELEMS (HID_FRAGS * 512)

// Fragment element mapping (CDNA5 ISA 7.12.2, 16-bit A/B 16x32):
//   lane L (0..31), elem e (0..15)  ->  K = kmap(e) + 8*(L>=16), row/col = L%16
//   kmap(e) = e + 8*(e>=8)   (K in {0..7,16..23})
// Each fragment stored linear: frag*512 + lane*16 + e   (1 KB per fragment)

// LDS offset (in __bf16 elems) of activation (row, col) in a tile with
// KB = K/32 fragments per 16-row strip.
__device__ __forceinline__ int lds_off(int row, int col, int KB) {
    int kr = col & 31;
    int hi = (kr >> 3) & 1;                      // which lane half holds this K
    int e  = (kr & 7) + ((kr >> 4) << 3);        // element index within lane
    int frag = (row >> 4) * KB + (col >> 5);
    return frag * 512 + ((row & 15) + (hi << 4)) * 16 + e;
}

// ---------------- weight swizzle (one-time per call, tiny) ----------------
__global__ void swizzle_weights(const float* __restrict__ Wfirst,
                                const float* __restrict__ Whid,
                                __bf16* __restrict__ ws) {
    int t = blockIdx.x * blockDim.x + threadIdx.x;   // one thread per (frag,lane)
    if (t >= TOT_FRAGS * 32) return;
    int fid  = t >> 5;
    int lane = t & 31;

    const float* W;
    int K, nb, kb;
    if (fid < L0_FRAGS) {
        W = Wfirst; K = FF_DIM;
        nb = fid >> 3; kb = fid & 7;
    } else {
        int g = fid - L0_FRAGS;
        int layer = g >> 9;                 // /512 frags per layer
        int r = g & 511;
        W = Whid + layer * HID * HID; K = HID;
        nb = r >> 4; kb = r & 15;
    }
    int n   = nb * 16 + (lane & 15);        // output-feature index (row of W)
    int khi = (lane >> 4) << 3;             // +8 for hi lane half
    __bf16* dst = ws + (size_t)fid * 512 + lane * 16;
#pragma unroll
    for (int e = 0; e < 16; ++e) {
        int k = kb * 32 + (e + ((e >= 8) ? 8 : 0)) + khi;
        dst[e] = (__bf16)W[n * K + k];
    }
}

// ---------------- one fused sine layer via WMMA ----------------
// Each wave computes a 64x64 output block: all 4 row-strips x 4 N-tiles.
// Per k-step: 4 A frags (LDS) + 4 B frags (global, L2) feed 16 independent
// WMMA chains -> 4x B-reuse, no intra-WG weight redundancy, no RAW stalls.
template <int KB>
__device__ __forceinline__ void sine_layer(const __bf16* __restrict__ hin,
                                           __bf16* __restrict__ hout,
                                           const __bf16* __restrict__ Wf,
                                           const float* __restrict__ bias,
                                           int wave, int lane) {
    const int nbase = wave * 4;            // this wave's 4 N-tiles

    v8f acc[4][4];                         // [m-strip][n-tile], 128 VGPRs
#pragma unroll
    for (int m = 0; m < 4; ++m)
#pragma unroll
        for (int j = 0; j < 4; ++j)
            acc[m][j] = (v8f){0.f,0.f,0.f,0.f,0.f,0.f,0.f,0.f};

    const __bf16* Wb = Wf + (((size_t)nbase * KB) << 9) + (lane << 4);
    const __bf16* Ab = hin + (lane << 4);

#pragma unroll
    for (int kb = 0; kb < KB; ++kb) {
        v16bf A[4], B[4];
#pragma unroll
        for (int m = 0; m < 4; ++m)
            A[m] = *(const v16bf*)(Ab + ((m * KB + kb) << 9));
#pragma unroll
        for (int j = 0; j < 4; ++j)
            B[j] = *(const v16bf*)(Wb + ((j * KB + kb) << 9));
#pragma unroll
        for (int m = 0; m < 4; ++m)
#pragma unroll
            for (int j = 0; j < 4; ++j)
                acc[m][j] = __builtin_amdgcn_wmma_f32_16x16x32_bf16(
                    false, A[m], false, B[j], (short)0, acc[m][j], false, false);
    }

    // D layout: lane -> col = lane%16 ; reg r -> row = r + 8*(lane>=16)
    const int nc = lane & 15;
    const int hi = lane >> 4;
#pragma unroll
    for (int j = 0; j < 4; ++j) {
        const int   nb  = nbase + j;
        const float bv  = bias[nb * 16 + nc];
        const int   col = nb * 16 + nc;
#pragma unroll
        for (int m = 0; m < 4; ++m) {
#pragma unroll
            for (int r = 0; r < 8; ++r) {
                int   row = m * 16 + r + (hi << 3);
                float y   = __sinf(OMEGA0 * (acc[m][j][r] + bv));
                hout[lds_off(row, col, 16)] = (__bf16)y;
            }
        }
    }
}

// ---------------- fully fused SIREN forward ----------------
__global__ void __launch_bounds__(THREADS)
siren_fused(const float* __restrict__ coords, const float* __restrict__ Bff,
            const float* __restrict__ b_first, const float* __restrict__ b_hidden,
            const float* __restrict__ Wout, const float* __restrict__ bout,
            const __bf16* __restrict__ ws, float* __restrict__ out) {
    __shared__ __align__(16) __bf16 hA[MTILE * HID];   // 64 KB
    __shared__ __align__(16) __bf16 hB[MTILE * HID];   // 64 KB

    const int tid  = threadIdx.x;
    const int lane = tid & 31;
    const int wave = tid >> 5;
    const int rowbase = blockIdx.x * MTILE;

    // ---- Fourier feature encode: h0 = [sin(2pi c@B), cos(2pi c@B)], KB=8 layout
    for (int idx = tid; idx < MTILE * MAP; idx += THREADS) {
        int row = idx >> 7;
        int j   = idx & 127;
        int gr  = rowbase + row;
        float c0 = coords[gr * 3 + 0];
        float c1 = coords[gr * 3 + 1];
        float c2 = coords[gr * 3 + 2];
        float p  = TWO_PI_F * (c0 * Bff[j] + c1 * Bff[MAP + j] + c2 * Bff[2 * MAP + j]);
        hA[lds_off(row, j,       8)] = (__bf16)__sinf(p);
        hA[lds_off(row, j + MAP, 8)] = (__bf16)__cosf(p);
    }
    __syncthreads();

    // ---- first sine layer: 256 -> 512
    sine_layer<8>(hA, hB, ws, b_first, wave, lane);
    __syncthreads();

    // ---- 5 hidden sine layers: 512 -> 512 (ping-pong hB <-> hA)
    for (int l = 0; l < NHID; ++l) {
        const __bf16* in  = (l & 1) ? hA : hB;
        __bf16*       dst = (l & 1) ? hB : hA;
        sine_layer<16>(in, dst,
                       ws + L0_ELEMS + (size_t)l * HID_ELEMS,
                       b_hidden + l * HID, wave, lane);
        __syncthreads();
    }
    // final activations in hA (after l=4 writes hA)

    // ---- output linear: 512 -> 3 (negligible cost, plain VALU)
    for (int t = tid; t < MTILE * OUTF; t += THREADS) {
        int row = t / 3;
        int o   = t - row * 3;
        float acc = bout[o];
        const int rbase = ((row >> 4) * 16) * 512 + ((row & 15) << 4);
#pragma unroll
        for (int kb = 0; kb < 16; ++kb) {
            int fb = rbase + (kb << 9);
#pragma unroll
            for (int kr = 0; kr < 32; ++kr) {
                int off = fb + (((kr >> 3) & 1) << 8) + (kr & 7) + ((kr >> 4) << 3);
                acc += (float)hA[off] * Wout[o * HID + (kb << 5) + kr];
            }
        }
        out[(size_t)(rowbase + row) * 3 + o] = acc;
    }
}

// ---------------- launch ----------------
extern "C" void kernel_launch(void* const* d_in, const int* in_sizes, int n_in,
                              void* d_out, int out_size, void* d_ws, size_t ws_size,
                              hipStream_t stream) {
    const float* coords  = (const float*)d_in[0];
    const float* Bff     = (const float*)d_in[1];
    const float* Wfirst  = (const float*)d_in[2];
    const float* bfirst  = (const float*)d_in[3];
    const float* Whid    = (const float*)d_in[4];
    const float* bhid    = (const float*)d_in[5];
    const float* Wout    = (const float*)d_in[6];
    const float* bout    = (const float*)d_in[7];
    __bf16* ws  = (__bf16*)d_ws;
    float*  out = (float*)d_out;

    // 1) swizzle weights f32 -> bf16 fragment layout (one thread per frag-lane)
    int prep = TOT_FRAGS * 32;
    swizzle_weights<<<(prep + 255) / 256, 256, 0, stream>>>(Wfirst, Whid, ws);

    // 2) fused SIREN forward: one workgroup per 64 rows
    siren_fused<<<L_TOTAL / MTILE, THREADS, 0, stream>>>(
        coords, Bff, bfirst, bhid, Wout, bout, ws, out);
}
